// AttentionHead_89824946029119
// MI455X (gfx1250) — matrix operand order repaired and checked by reading, not compiled, not run on hardware
//
#include <hip/hip_runtime.h>

// Problem constants (from reference)
constexpr int kB = 8;
constexpr int kS = 2048;
constexpr int kW = 1024;
constexpr int kH = 64;

typedef __attribute__((ext_vector_type(16))) _Float16 v16h;
typedef __attribute__((ext_vector_type(8)))  _Float16 v8h;
typedef __attribute__((ext_vector_type(4)))  _Float16 v4h;
typedef __attribute__((ext_vector_type(8)))  float    v8f;

// LDS strides in halfs; multiples of 8 halfs (16B) so ds_load_b128 stays aligned
#define LD32 40   // rows holding 32 halfs (80B, 16B-aligned, banks staggered)
#define LD64 72   // rows holding 64 halfs (144B, 16B-aligned)

__device__ __forceinline__ v16h cat16(v8h lo, v8h hi) {
  return __builtin_shufflevector(lo, hi, 0,1,2,3,4,5,6,7,8,9,10,11,12,13,14,15);
}

// A-matrix fragment (16x32 f16, ISA 7.12.2): lane half hf holds
// K = [hf*8, hf*8+8) and K = [16+hf*8, 16+hf*8+8)  -> two b128 LDS loads.
__device__ __forceinline__ v16h load_afrag(const _Float16* rowp, int hf) {
  v8h lo = *(const v8h*)(rowp + hf * 8);
  v8h hi = *(const v8h*)(rowp + 16 + hf * 8);
  return cat16(lo, hi);
}
// B-matrix fragment (32x16 f16) from an [N][K]-layout tile: lane half hf holds
// K = [hf*8, hf*8+8) and K = [16+hf*8, 16+hf*8+8) of row N=lane%16.
__device__ __forceinline__ v16h load_bfrag(const _Float16* rowp, int hf) {
  v8h lo = *(const v8h*)(rowp + hf * 8);
  v8h hi = *(const v8h*)(rowp + 16 + hf * 8);
  return cat16(lo, hi);
}

#define WMMA_F16(A, Bm, C) \
  __builtin_amdgcn_wmma_f32_16x16x32_f16(false, (A), false, (Bm), (short)0, (C), false, false)

// ---------------------------------------------------------------------------
// Fused QKV projection: one pass over x, 12 WMMAs per 32-wide K step per wave.
// Weight tiles staged TRANSPOSED ([n][k]) so B fragments are contiguous.
// Fragment loads batched (6 at a time) so dscnt waits are partial, not 0.
// ---------------------------------------------------------------------------
__global__ __launch_bounds__(128)
void attn_proj_kernel(const float* __restrict__ x,
                      const float* __restrict__ Wq, const float* __restrict__ biasq,
                      const float* __restrict__ Wk, const float* __restrict__ biask,
                      const float* __restrict__ Wv, const float* __restrict__ biasv,
                      _Float16* __restrict__ q16, _Float16* __restrict__ k16,
                      _Float16* __restrict__ v16)
{
  __shared__ alignas(16) _Float16 xs[64][LD32];     // [row][k]
  __shared__ alignas(16) _Float16 wqT[64][LD32];    // [n][k]
  __shared__ alignas(16) _Float16 wkT[64][LD32];
  __shared__ alignas(16) _Float16 wvT[64][LD32];

  const int tid  = threadIdx.x;
  const int wv   = tid >> 5;
  const int lane = tid & 31;
  const int hf   = lane >> 4;
  const int lh   = lane & 15;
  const int row0 = blockIdx.x * 64;

  v8f accq[4], acck[4], accv[4];
  #pragma unroll
  for (int nt = 0; nt < 4; ++nt) { accq[nt] = (v8f)0.f; acck[nt] = (v8f)0.f; accv[nt] = (v8f)0.f; }

  for (int k0 = 0; k0 < kW; k0 += 32) {
    __syncthreads();
    // stage x tile 64x32 via float4 loads + packed f16 stores
    for (int i = tid; i < 512; i += 128) {           // 64*32/4 float4 quads
      int r = i >> 3, c4 = (i & 7) * 4;
      float4 f = *(const float4*)(x + (size_t)(row0 + r) * kW + k0 + c4);
      v4h h = { (_Float16)f.x, (_Float16)f.y, (_Float16)f.z, (_Float16)f.w };
      *(v4h*)&xs[r][c4] = h;
    }
    // stage weight tiles 32x64 transposed -> [n][k]
    for (int i = tid; i < 512; i += 128) {           // 32*64/4
      int r = i >> 4, c4 = (i & 15) * 4;             // r = k row, c4 = n col
      size_t gi = (size_t)(k0 + r) * kH + c4;
      float4 fq = *(const float4*)(Wq + gi);
      float4 fk = *(const float4*)(Wk + gi);
      float4 fv = *(const float4*)(Wv + gi);
      wqT[c4+0][r] = (_Float16)fq.x; wqT[c4+1][r] = (_Float16)fq.y;
      wqT[c4+2][r] = (_Float16)fq.z; wqT[c4+3][r] = (_Float16)fq.w;
      wkT[c4+0][r] = (_Float16)fk.x; wkT[c4+1][r] = (_Float16)fk.y;
      wkT[c4+2][r] = (_Float16)fk.z; wkT[c4+3][r] = (_Float16)fk.w;
      wvT[c4+0][r] = (_Float16)fv.x; wvT[c4+1][r] = (_Float16)fv.y;
      wvT[c4+2][r] = (_Float16)fv.z; wvT[c4+3][r] = (_Float16)fv.w;
    }
    __syncthreads();

    v16h a = load_afrag(&xs[wv * 16 + lh][0], hf);
    #pragma unroll
    for (int g = 0; g < 2; ++g) {                    // two N-tile pairs
      int n0 = (g * 2 + 0) * 16 + lh;
      int n1 = (g * 2 + 1) * 16 + lh;
      // batch 6 fragment loads, then 6 WMMAs -> partial dscnt waits
      v16h bq0 = load_bfrag(&wqT[n0][0], hf);
      v16h bk0 = load_bfrag(&wkT[n0][0], hf);
      v16h bv0 = load_bfrag(&wvT[n0][0], hf);
      v16h bq1 = load_bfrag(&wqT[n1][0], hf);
      v16h bk1 = load_bfrag(&wkT[n1][0], hf);
      v16h bv1 = load_bfrag(&wvT[n1][0], hf);
      accq[g*2+0] = WMMA_F16(a, bq0, accq[g*2+0]);
      acck[g*2+0] = WMMA_F16(a, bk0, acck[g*2+0]);
      accv[g*2+0] = WMMA_F16(a, bv0, accv[g*2+0]);
      accq[g*2+1] = WMMA_F16(a, bq1, accq[g*2+1]);
      acck[g*2+1] = WMMA_F16(a, bk1, acck[g*2+1]);
      accv[g*2+1] = WMMA_F16(a, bv1, accv[g*2+1]);
    }
  }

  // epilogue: +bias, convert, store f16 row-major [B*S, H]
  #pragma unroll
  for (int nt = 0; nt < 4; ++nt) {
    int n = nt * 16 + lh;
    float bq_s = biasq[n], bk_s = biask[n], bv_s = biasv[n];
    #pragma unroll
    for (int r = 0; r < 8; ++r) {
      size_t row = (size_t)(row0 + wv * 16 + r + hf * 8);
      q16[row * kH + n] = (_Float16)(accq[nt][r] + bq_s);
      k16[row * kH + n] = (_Float16)(acck[nt][r] + bk_s);
      v16[row * kH + n] = (_Float16)(accv[nt][r] + bv_s);
    }
  }
}

// ---------------------------------------------------------------------------
// Flash attention: 64 queries per block (16/wave), 32-key blocks through LDS.
// K tile kept [key][h] (natural [N][K] for the scores B fragments);
// V tile staged TRANSPOSED [h][key] so the P.V B fragments are contiguous.
// ---------------------------------------------------------------------------
__global__ __launch_bounds__(128)
void attn_flash_kernel(const _Float16* __restrict__ q16,
                       const _Float16* __restrict__ k16,
                       const _Float16* __restrict__ v16,
                       const int* __restrict__ attn_mask,
                       float* __restrict__ out)
{
  __shared__ alignas(16) _Float16 qs[64][LD64];      // [q][h]
  __shared__ alignas(16) _Float16 ks[32][LD64];      // [key][h]
  __shared__ alignas(16) _Float16 vsT[64][LD32];     // [h][key]
  __shared__ alignas(16) _Float16 ps[4][16][LD32];   // per-wave P [q][key]

  const int tid  = threadIdx.x;
  const int wv   = tid >> 5;
  const int lane = tid & 31;
  const int hf   = lane >> 4;
  const int lh   = lane & 15;
  const int b    = blockIdx.y;
  const int q0   = blockIdx.x * 64;
  const size_t baseQ = ((size_t)b * kS + q0) * kH;

  // stage Q 64x64 with b128 copies, build persistent A fragments
  for (int i = tid; i < 512; i += 128) {             // 64*64/8
    int r = i >> 3, c8 = (i & 7) * 8;
    *(v8h*)&qs[r][c8] = *(const v8h*)(q16 + baseQ + (size_t)r * kH + c8);
  }
  __syncthreads();
  v16h aq0 = load_afrag(&qs[wv * 16 + lh][0],  hf);
  v16h aq1 = load_afrag(&qs[wv * 16 + lh][32], hf);

  const float scale = 0.125f;  // 1/sqrt(64)
  float mrow[8], lrow[8];
  v8f acc[4];
  #pragma unroll
  for (int r = 0; r < 8; ++r) { mrow[r] = -1.0e30f; lrow[r] = 0.f; }
  #pragma unroll
  for (int nt = 0; nt < 4; ++nt) acc[nt] = (v8f)0.f;

  for (int kb = 0; kb < kS; kb += 32) {
    __syncthreads();
    const size_t baseK = ((size_t)b * kS + kb) * kH;
    for (int i = tid; i < 256; i += 128) {           // 32*64/8
      int r = i >> 3, c8 = (i & 7) * 8;
      *(v8h*)&ks[r][c8] = *(const v8h*)(k16 + baseK + (size_t)r * kH + c8);
      v8h vv = *(const v8h*)(v16 + baseK + (size_t)r * kH + c8);
      #pragma unroll
      for (int e = 0; e < 8; ++e) vsT[c8 + e][r] = vv[e];   // transpose
    }
    __syncthreads();
    if (kb + 32 < kS) {                              // global_prefetch_b8 next block
      __builtin_prefetch(&k16[baseK + (size_t)32 * kH + (tid << 4)], 0, 0);
      __builtin_prefetch(&v16[baseK + (size_t)32 * kH + (tid << 4)], 0, 0);
    }

    // scores: S = Q.K^T  -- batch all 4 K fragments, then 4 WMMAs
    v16h bk00 = load_bfrag(&ks[lh][0],       hf);
    v16h bk01 = load_bfrag(&ks[lh][32],      hf);
    v16h bk10 = load_bfrag(&ks[16 + lh][0],  hf);
    v16h bk11 = load_bfrag(&ks[16 + lh][32], hf);
    v8f sc[2];
    sc[0] = (v8f)0.f;  sc[1] = (v8f)0.f;
    sc[0] = WMMA_F16(aq0, bk00, sc[0]);
    sc[1] = WMMA_F16(aq0, bk10, sc[1]);
    sc[0] = WMMA_F16(aq1, bk01, sc[0]);
    sc[1] = WMMA_F16(aq1, bk11, sc[1]);

    // scale + key-padding mask
    int m0 = attn_mask[(size_t)b * kS + kb + lh];
    int m1 = attn_mask[(size_t)b * kS + kb + 16 + lh];
    float ss0[8], ss1[8];
    #pragma unroll
    for (int r = 0; r < 8; ++r) {
      ss0[r] = m0 ? sc[0][r] * scale : -1.0e30f;
      ss1[r] = m1 ? sc[1][r] * scale : -1.0e30f;
    }

    // online softmax; row = r + 8*hf, N striped across the 16 lanes of a half
    #pragma unroll
    for (int r = 0; r < 8; ++r) {
      float rm = fmaxf(ss0[r], ss1[r]);
      #pragma unroll
      for (int mm = 1; mm < 16; mm <<= 1) rm = fmaxf(rm, __shfl_xor(rm, mm, 32));
      float mnew  = fmaxf(mrow[r], rm);
      float alpha = __expf(mrow[r] - mnew);
      mrow[r] = mnew;
      float p0 = __expf(ss0[r] - mnew);
      float p1 = __expf(ss1[r] - mnew);
      float rs = p0 + p1;
      #pragma unroll
      for (int mm = 1; mm < 16; mm <<= 1) rs += __shfl_xor(rs, mm, 32);
      lrow[r] = lrow[r] * alpha + rs;
      acc[0][r] *= alpha; acc[1][r] *= alpha; acc[2][r] *= alpha; acc[3][r] *= alpha;
      ps[wv][r + hf * 8][lh]      = (_Float16)p0;
      ps[wv][r + hf * 8][16 + lh] = (_Float16)p1;
    }

    // O += P.V : batch A + 4 V fragments, then 4 WMMAs
    v16h ap  = load_afrag(&ps[wv][lh][0], hf);
    v16h bv0 = load_bfrag(&vsT[lh][0],      hf);
    v16h bv1 = load_bfrag(&vsT[16 + lh][0], hf);
    v16h bv2 = load_bfrag(&vsT[32 + lh][0], hf);
    v16h bv3 = load_bfrag(&vsT[48 + lh][0], hf);
    acc[0] = WMMA_F16(ap, bv0, acc[0]);
    acc[1] = WMMA_F16(ap, bv1, acc[1]);
    acc[2] = WMMA_F16(ap, bv2, acc[2]);
    acc[3] = WMMA_F16(ap, bv3, acc[3]);
  }

  // epilogue: O / l -> fp32 out [B,S,H]
  #pragma unroll
  for (int r = 0; r < 8; ++r) {
    float inv = 1.0f / lrow[r];
    size_t row = (size_t)b * kS + q0 + wv * 16 + r + hf * 8;
    #pragma unroll
    for (int nt = 0; nt < 4; ++nt)
      out[row * kH + nt * 16 + lh] = acc[nt][r] * inv;
  }
}

// ---------------------------------------------------------------------------
extern "C" void kernel_launch(void* const* d_in, const int* in_sizes, int n_in,
                              void* d_out, int out_size, void* d_ws, size_t ws_size,
                              hipStream_t stream) {
  (void)in_sizes; (void)n_in; (void)out_size; (void)ws_size;
  const float* x     = (const float*)d_in[0];
  const int*   amask = (const int*)d_in[1];
  const float* Wq    = (const float*)d_in[2];
  const float* bq    = (const float*)d_in[3];
  const float* Wk    = (const float*)d_in[4];
  const float* bk    = (const float*)d_in[5];
  const float* Wv    = (const float*)d_in[6];
  const float* bv    = (const float*)d_in[7];

  _Float16* q16 = (_Float16*)d_ws;                   // 3 x 2MB f16 staging
  _Float16* k16 = q16 + (size_t)kB * kS * kH;
  _Float16* v16 = k16 + (size_t)kB * kS * kH;

  dim3 gProj((kB * kS) / 64);
  attn_proj_kernel<<<gProj, 128, 0, stream>>>(x, Wq, bq, Wk, bk, Wv, bv, q16, k16, v16);

  dim3 gAttn(kS / 64, kB);
  attn_flash_kernel<<<gAttn, 128, 0, stream>>>(q16, k16, v16, amask, (float*)d_out);
}
